// ViTBackbone_3874060501000
// MI455X (gfx1250) — compile-verified, matrix operand order checked by reading, not run on hardware
//
#include <hip/hip_runtime.h>
#include <hip/hip_bf16.h>
#include <stdint.h>

// ---------------------------------------------------------------------------
// ViT-Base forward for MI455X (gfx1250): all GEMMs via v_wmma_f32_16x16x32_bf16
// (wave32, 16x16x32 bf16 -> f32 accum). fp32 residual/LN/softmax on VALU.
// Compute-bound (~2.35 TFLOP vs <1GB traffic @23.3TB/s) => WMMA path, with
// async global->LDS copies (ASYNCcnt) + double-buffered LDS pipeline.
// ---------------------------------------------------------------------------

typedef __attribute__((ext_vector_type(16))) __bf16 v16bf;
typedef __attribute__((ext_vector_type(8)))  __bf16 v8bf;
typedef __attribute__((ext_vector_type(8)))  float  v8f;

#define BM 128
#define BN 128
#define BK 32
#define LDSP 48   // padded row stride (bf16 elems), 96B = 6*16B -> aligned b128

#define USE_ASYNC 1

#if USE_ASYNC
#if defined(__has_builtin) && __has_builtin(__builtin_amdgcn_s_wait_asynccnt)
#define WAIT_ASYNC() __builtin_amdgcn_s_wait_asynccnt(0)
#else
#define WAIT_ASYNC() asm volatile("s_wait_asynccnt 0x0" ::: "memory")
#endif
#else
#define WAIT_ASYNC() ((void)0)
#endif

// ------------------------------ model dims ---------------------------------
#define Bsz 64
#define Hh  12
#define Eo  768
#define Dd  64
#define Nt  197
#define Ll  12
#define Ff  3072
#define NP  196
#define Ttok (Bsz * Nt)     // 12608
#define Tpat (Bsz * NP)     // 12544

// --------------------- tile staging (global -> LDS) ------------------------
__device__ __forceinline__ void stage_tile(
    const __bf16* __restrict__ A, const __bf16* __restrict__ Bm,
    __bf16 (*As)[LDSP], __bf16 (*Bs)[LDSP],
    int tid, int bm0, int bn0, int k0,
    int M, int N, int K, int lda, int ldb, bool aVec, bool bVec)
{
    // A tile (BMxBK) row-major: async copy global->LDS, no VGPR round-trip
    for (int c = tid; c < (BM * BK / 8); c += 256) {
        int row = c >> 2, cc = c & 3;
        int gr = bm0 + row, kb = k0 + cc * 8;
        __bf16* dst = &As[row][cc * 8];
        if (gr < M && (kb + 8) <= K && aVec) {
#if USE_ASYNC
            asm volatile("global_load_async_to_lds_b128 %0, %1, off"
                         :: "v"((unsigned)(uintptr_t)dst),
                            "v"(A + (long long)gr * lda + kb)
                         : "memory");
#else
            *(v8bf*)dst = *(const v8bf*)(A + (long long)gr * lda + kb);
#endif
        } else {
            for (int j = 0; j < 8; ++j)
                dst[j] = (gr < M && (kb + j) < K)
                       ? A[(long long)gr * lda + kb + j] : (__bf16)0.0f;
        }
    }
    // B tile (BKxBN) staged transposed (N-major) for fragment-friendly reads
    for (int c = tid; c < (BK * BN / 8); c += 256) {
        int krow = c >> 4, nc = c & 15;
        int gk = k0 + krow, gn = bn0 + nc * 8;
        v8bf v;
        if (gk < K && (gn + 8) <= N && bVec) {
            v = *(const v8bf*)(Bm + (long long)gk * ldb + gn);
        } else {
            for (int j = 0; j < 8; ++j)
                v[j] = (gk < K && (gn + j) < N)
                     ? Bm[(long long)gk * ldb + gn + j] : (__bf16)0.0f;
        }
        for (int j = 0; j < 8; ++j) Bs[nc * 8 + j][krow] = v[j];
    }
}

// ======================= generic batched WMMA GEMM =========================
// C[z] = act( A[z](MxK) * B[z](KxN) + bias ) [+ Res] ; A,B bf16 row-major.
// Per-z offsets: A += z*sA ; B += z*sB ; C/Res += (z/Hdiv)*sCo + (z%Hdiv)*sCi.
// flags bit0: exact-erf GELU.
__global__ __launch_bounds__(256) void k_gemm(
    const __bf16* __restrict__ A, const __bf16* __restrict__ Bm,
    const float* __restrict__ bias, const float* __restrict__ Res,
    float* __restrict__ Cf, __bf16* __restrict__ Cbf,
    int M, int N, int K, int lda, int ldb, int ldc,
    long long sA, long long sB, int Hdiv, long long sCo, long long sCi,
    int flags)
{
    __shared__ __bf16 As[2][BM][LDSP];
    __shared__ __bf16 Bs[2][BN][LDSP];

    const int tid  = threadIdx.x;
    const int lane = tid & 31;
    const int wave = tid >> 5;
    const int lr   = lane & 15;
    const int half = lane >> 4;
    const int kb0  = half * 8;

    const int z   = blockIdx.z;
    const int bm0 = blockIdx.y * BM;
    const int bn0 = blockIdx.x * BN;

    A  += (long long)z * sA;
    Bm += (long long)z * sB;
    const long long cOff = (long long)(z / Hdiv) * sCo + (long long)(z % Hdiv) * sCi;

    const int wm = (wave >> 2) * 64;   // 0 / 64
    const int wn = (wave & 3) * 32;    // 0 / 32 / 64 / 96

    const bool aVec = ((lda & 7) == 0);
    const bool bVec = ((ldb & 7) == 0);

    v8f acc[4][2] = {};

    const int nk = (K + BK - 1) / BK;

    // prologue: stage tile 0
    stage_tile(A, Bm, As[0], Bs[0], tid, bm0, bn0, 0, M, N, K, lda, ldb, aVec, bVec);
    WAIT_ASYNC();
    __syncthreads();

    int p = 0;
    for (int t = 0; t < nk; ++t) {
        // stage next tile into the alternate buffer (async, in flight during WMMA)
        if (t + 1 < nk)
            stage_tile(A, Bm, As[p ^ 1], Bs[p ^ 1], tid, bm0, bn0, (t + 1) * BK,
                       M, N, K, lda, ldb, aVec, bVec);
        // prefetch tile t+2 toward L2 (global_prefetch_b8)
        if (t + 2 < nk) {
            int k2 = (t + 2) * BK;
            int pr = bm0 + (tid & 127);
            if (pr < M) __builtin_prefetch(A + (long long)pr * lda + k2, 0, 3);
            if (tid < BK && (k2 + tid) < K)
                __builtin_prefetch(Bm + (long long)(k2 + tid) * ldb + bn0, 0, 3);
        }

        // ---- gather fragments (ISA 16-bit A/B layout) & 8 WMMAs ----
        v16bf afr[4], bfr[2];
        for (int i = 0; i < 4; ++i) {
            int m = wm + i * 16 + lr;
            v8bf lo = *(const v8bf*)&As[p][m][kb0];
            v8bf hi = *(const v8bf*)&As[p][m][kb0 + 16];
            afr[i] = __builtin_shufflevector(lo, hi, 0,1,2,3,4,5,6,7,8,9,10,11,12,13,14,15);
        }
        for (int j = 0; j < 2; ++j) {
            int n = wn + j * 16 + lr;
            v8bf lo = *(const v8bf*)&Bs[p][n][kb0];
            v8bf hi = *(const v8bf*)&Bs[p][n][kb0 + 16];
            bfr[j] = __builtin_shufflevector(lo, hi, 0,1,2,3,4,5,6,7,8,9,10,11,12,13,14,15);
        }
        for (int i = 0; i < 4; ++i)
            for (int j = 0; j < 2; ++j)
                acc[i][j] = __builtin_amdgcn_wmma_f32_16x16x32_bf16(
                    false, afr[i], false, bfr[j], (short)0, acc[i][j], false, false);

        if (t + 1 < nk) {
            WAIT_ASYNC();      // my async LDS writes landed
            __syncthreads();   // everyone done reading old buffer / writing new
        }
        p ^= 1;
    }

    // ------------------------------ epilogue -------------------------------
    for (int i = 0; i < 4; ++i) {
        for (int j = 0; j < 2; ++j) {
            int col = bn0 + wn + j * 16 + lr;
            if (col >= N) continue;
            float bv = bias ? bias[col] : 0.0f;
            for (int r = 0; r < 8; ++r) {
                int row = bm0 + wm + i * 16 + half * 8 + r;
                if (row >= M) continue;
                float v = acc[i][j][r] + bv;
                if (flags & 1) v = 0.5f * v * (1.0f + erff(v * 0.70710678118654752f));
                long long off = cOff + (long long)row * ldc + col;
                if (Res) v += Res[off];
                if (Cf)  Cf[off]  = v;
                if (Cbf) Cbf[off] = (__bf16)v;
            }
        }
    }
}

// =========================== elementwise kernels ===========================
__global__ void k_cvt_bf16(__bf16* __restrict__ d, const float* __restrict__ s, long long n) {
    long long i = (long long)blockIdx.x * blockDim.x + threadIdx.x;
    if (i < n) d[i] = (__bf16)s[i];
}

// qkv_w (L,H,E,192) -> per layer [E, H*192] row-major
__global__ void k_cvt_qkvw(__bf16* __restrict__ d, const float* __restrict__ s) {
    long long n = (long long)Ll * Hh * Eo * 192;
    long long i = (long long)blockIdx.x * blockDim.x + threadIdx.x;
    if (i >= n) return;
    int j = (int)(i % 192);
    int e = (int)((i / 192) % Eo);
    int h = (int)((i / (192LL * Eo)) % Hh);
    long long l = i / (192LL * Eo * Hh);
    d[(l * Eo + e) * (3 * Eo) + h * 192 + j] = (__bf16)s[i];
}

// conv_w (E, C*P*P) -> [C*P*P, E]
__global__ void k_cvt_convw(__bf16* __restrict__ d, const float* __restrict__ s) {
    long long i = (long long)blockIdx.x * blockDim.x + threadIdx.x;
    if (i >= (long long)Eo * Eo) return;
    int e = (int)(i / Eo), k = (int)(i % Eo);
    d[(long long)k * Eo + e] = (__bf16)s[i];
}

// im2col: patches[b*196+p][c*256+ph*16+pw] (bf16)
__global__ void k_patchify(__bf16* __restrict__ d, const float* __restrict__ x) {
    long long i = (long long)blockIdx.x * blockDim.x + threadIdx.x;
    if (i >= (long long)Tpat * Eo) return;
    int col = (int)(i % Eo);
    long long row = i / Eo;
    int b = (int)(row / NP), p = (int)(row % NP);
    int py = p / 14, px = p % 14;
    int c = col >> 8, rem = col & 255;
    int ph = rem >> 4, pw = rem & 15;
    d[i] = (__bf16)x[(((long long)b * 3 + c) * 224 + (py * 16 + ph)) * 224 + (px * 16 + pw)];
}

__global__ void k_assemble(float* __restrict__ h, const float* __restrict__ conv_out,
                           const float* __restrict__ cls, const float* __restrict__ pos) {
    long long i = (long long)blockIdx.x * blockDim.x + threadIdx.x;
    if (i >= (long long)Ttok * Eo) return;
    int e = (int)(i % Eo);
    int n = (int)((i / Eo) % Nt);
    long long b = i / ((long long)Eo * Nt);
    float v = pos[(long long)n * Eo + e];
    if (n == 0) v += cls[e];
    else        v += conv_out[((long long)b * NP + (n - 1)) * Eo + e];
    h[i] = v;
}

// LayerNorm over E=768, fused bf16 output (one block per token)
__global__ __launch_bounds__(256) void k_layernorm(const float* __restrict__ X,
    const float* __restrict__ g, const float* __restrict__ b, __bf16* __restrict__ Y)
{
    long long row = blockIdx.x;
    const float* x = X + row * Eo;
    int tid = threadIdx.x;
    float s = 0.f, q = 0.f;
    for (int e = tid; e < Eo; e += 256) { float v = x[e]; s += v; q += v * v; }
    for (int o = 16; o; o >>= 1) { s += __shfl_xor(s, o, 32); q += __shfl_xor(q, o, 32); }
    __shared__ float rs[8], rq[8];
    if ((tid & 31) == 0) { rs[tid >> 5] = s; rq[tid >> 5] = q; }
    __syncthreads();
    if (tid == 0) {
        float ts = 0.f, tq = 0.f;
        for (int i = 0; i < 8; ++i) { ts += rs[i]; tq += rq[i]; }
        float m = ts * (1.0f / Eo);
        float var = tq * (1.0f / Eo) - m * m;
        rs[0] = m; rq[0] = rsqrtf(var + 1e-5f);
    }
    __syncthreads();
    float m = rs[0], r = rq[0];
    for (int e = tid; e < Eo; e += 256)
        Y[row * Eo + e] = (__bf16)((x[e] - m) * r * g[e] + b[e]);
}

// de-interleave qkv (torch view(...,Dh,3).unbind): j = 3*d + {0,1,2}
__global__ void k_repack_qkv(const float* __restrict__ qkv,
    __bf16* __restrict__ Qb, __bf16* __restrict__ Kt, __bf16* __restrict__ Vb)
{
    long long n = (long long)Bsz * Hh * Nt * Dd;
    long long i = (long long)blockIdx.x * blockDim.x + threadIdx.x;
    if (i >= n) return;
    int d  = (int)(i % Dd);
    int nn = (int)((i / Dd) % Nt);
    int h  = (int)((i / ((long long)Dd * Nt)) % Hh);
    long long b = i / ((long long)Dd * Nt * Hh);
    const float* q = qkv + ((long long)(b * Nt + nn)) * (3 * Eo) + h * 192 + d * 3;
    long long bh = b * Hh + h;
    Qb[(bh * Nt + nn) * Dd + d] = (__bf16)(q[0] * 0.125f);  // 1/sqrt(64) folded
    Kt[(bh * Dd + d) * Nt + nn] = (__bf16)q[1];
    Vb[(bh * Nt + nn) * Dd + d] = (__bf16)q[2];
}

// softmax over rows of length 197, bf16 output (one block of 128 per row)
__global__ __launch_bounds__(128) void k_softmax(const float* __restrict__ S, __bf16* __restrict__ P) {
    long long row = blockIdx.x;
    const float* s = S + row * Nt;
    __shared__ float xb[Nt];
    __shared__ float r1[4], r2[4];
    int tid = threadIdx.x;
    float mx = -3.0e38f;
    for (int i = tid; i < Nt; i += 128) { float v = s[i]; xb[i] = v; mx = fmaxf(mx, v); }
    for (int o = 16; o; o >>= 1) mx = fmaxf(mx, __shfl_xor(mx, o, 32));
    if ((tid & 31) == 0) r1[tid >> 5] = mx;
    __syncthreads();
    mx = fmaxf(fmaxf(r1[0], r1[1]), fmaxf(r1[2], r1[3]));
    float sum = 0.f;
    for (int i = tid; i < Nt; i += 128) { float e = expf(xb[i] - mx); xb[i] = e; sum += e; }
    for (int o = 16; o; o >>= 1) sum += __shfl_xor(sum, o, 32);
    if ((tid & 31) == 0) r2[tid >> 5] = sum;
    __syncthreads();
    sum = r2[0] + r2[1] + r2[2] + r2[3];
    float inv = 1.0f / sum;
    for (int i = tid; i < Nt; i += 128) P[row * Nt + i] = (__bf16)(xb[i] * inv);
}

// ================================= host ====================================
static inline int nblk(long long n) { return (int)((n + 255) / 256); }

extern "C" void kernel_launch(void* const* d_in, const int* in_sizes, int n_in,
                              void* d_out, int out_size, void* d_ws, size_t ws_size,
                              hipStream_t stream) {
    (void)in_sizes; (void)n_in; (void)out_size; (void)ws_size;
    const float* x      = (const float*)d_in[0];
    const float* conv_w = (const float*)d_in[1];
    const float* conv_b = (const float*)d_in[2];
    const float* cls_e  = (const float*)d_in[3];
    const float* pos_e  = (const float*)d_in[4];
    const float* ln1_s  = (const float*)d_in[5];
    const float* ln1_b  = (const float*)d_in[6];
    const float* qkv_w  = (const float*)d_in[7];
    const float* qkv_b  = (const float*)d_in[8];
    const float* proj_w = (const float*)d_in[9];
    const float* proj_b = (const float*)d_in[10];
    const float* ln2_s  = (const float*)d_in[11];
    const float* ln2_b  = (const float*)d_in[12];
    const float* fc1_w  = (const float*)d_in[13];
    const float* fc1_b  = (const float*)d_in[14];
    const float* fc2_w  = (const float*)d_in[15];
    const float* fc2_b  = (const float*)d_in[16];

    char* ws = (char*)d_ws;
    size_t off = 0;
    auto take = [&](size_t bytes) -> char* {
        off = (off + 255) & ~(size_t)255;
        char* p = ws + off;
        off += bytes;
        return p;
    };

    __bf16* wqkv  = (__bf16*)take((size_t)Ll * Eo * 3 * Eo * 2);
    __bf16* wproj = (__bf16*)take((size_t)Ll * Eo * Eo * 2);
    __bf16* wfc1  = (__bf16*)take((size_t)Ll * Eo * Ff * 2);
    __bf16* wfc2  = (__bf16*)take((size_t)Ll * Ff * Eo * 2);
    __bf16* wconv = (__bf16*)take((size_t)Eo * Eo * 2);
    // qkvf region also hosts patchify+conv_out (dead before qkvf first use)
    char*   bigR  = take((size_t)Ttok * 3 * Eo * 4);
    float*  qkvf  = (float*)bigR;
    __bf16* patches  = (__bf16*)bigR;
    float*  conv_out = (float*)(bigR + (((size_t)Tpat * Eo * 2 + 255) & ~(size_t)255));
    float*  hX    = (float*)take((size_t)Ttok * Eo * 4);
    float*  hY    = (float*)take((size_t)Ttok * Eo * 4);
    __bf16* lnbf  = (__bf16*)take((size_t)Ttok * Eo * 2);
    __bf16* Qb    = (__bf16*)take((size_t)Bsz * Hh * Nt * Dd * 2);
    __bf16* Kt    = (__bf16*)take((size_t)Bsz * Hh * Dd * Nt * 2);
    __bf16* Vb    = (__bf16*)take((size_t)Bsz * Hh * Nt * Dd * 2);
    float*  Smat  = (float*)take((size_t)Bsz * Hh * Nt * Nt * 4);
    __bf16* Pmat  = (__bf16*)take((size_t)Bsz * Hh * Nt * Nt * 2);
    __bf16* attbf = (__bf16*)take((size_t)Ttok * Eo * 2);
    __bf16* mlpbf = (__bf16*)take((size_t)Ttok * Ff * 2);

    auto gemm = [&](const __bf16* A, const __bf16* Bm, const float* bias, const float* Res,
                    float* Cf, __bf16* Cbf, int M, int N, int K, int lda, int ldb, int ldc,
                    long long sA, long long sB, int Hdiv, long long sCo, long long sCi,
                    int flags, int batches) {
        dim3 g((N + BN - 1) / BN, (M + BM - 1) / BM, batches);
        k_gemm<<<g, dim3(256), 0, stream>>>(A, Bm, bias, Res, Cf, Cbf, M, N, K,
                                            lda, ldb, ldc, sA, sB, Hdiv, sCo, sCi, flags);
    };

    // ---- weight conversion (every call; deterministic) ----
    k_cvt_qkvw <<<nblk((long long)Ll * Hh * Eo * 192), 256, 0, stream>>>(wqkv, qkv_w);
    k_cvt_bf16 <<<nblk((long long)Ll * Eo * Eo), 256, 0, stream>>>(wproj, proj_w, (long long)Ll * Eo * Eo);
    k_cvt_bf16 <<<nblk((long long)Ll * Eo * Ff), 256, 0, stream>>>(wfc1, fc1_w, (long long)Ll * Eo * Ff);
    k_cvt_bf16 <<<nblk((long long)Ll * Ff * Eo), 256, 0, stream>>>(wfc2, fc2_w, (long long)Ll * Ff * Eo);
    k_cvt_convw<<<nblk((long long)Eo * Eo), 256, 0, stream>>>(wconv, conv_w);

    // ---- patch embedding ----
    k_patchify<<<nblk((long long)Tpat * Eo), 256, 0, stream>>>(patches, x);
    gemm(patches, wconv, conv_b, nullptr, conv_out, nullptr,
         Tpat, Eo, Eo, Eo, Eo, Eo, 0, 0, 1, 0, 0, 0, 1);
    k_assemble<<<nblk((long long)Ttok * Eo), 256, 0, stream>>>(hX, conv_out, cls_e, pos_e);

    // ---- transformer layers ----
    for (int l = 0; l < Ll; ++l) {
        k_layernorm<<<Ttok, 256, 0, stream>>>(hX, ln1_s + (size_t)l * Eo, ln1_b + (size_t)l * Eo, lnbf);

        gemm(lnbf, wqkv + (size_t)l * Eo * 3 * Eo, qkv_b + (size_t)l * 3 * Eo, nullptr,
             qkvf, nullptr, Ttok, 3 * Eo, Eo, Eo, 3 * Eo, 3 * Eo, 0, 0, 1, 0, 0, 0, 1);

        k_repack_qkv<<<nblk((long long)Bsz * Hh * Nt * Dd), 256, 0, stream>>>(qkvf, Qb, Kt, Vb);

        // S = (Q/8) K^T   : 768 batches of [197x64]x[64x197]
        gemm(Qb, Kt, nullptr, nullptr, Smat, nullptr,
             Nt, Nt, Dd, Dd, Nt, Nt,
             (long long)Nt * Dd, (long long)Dd * Nt, 1, (long long)Nt * Nt, 0, 0,
             Bsz * Hh);

        k_softmax<<<Bsz * Hh * Nt, 128, 0, stream>>>(Smat, Pmat);

        // att = P V -> bf16 [T, E] with column block h*64
        gemm(Pmat, Vb, nullptr, nullptr, nullptr, attbf,
             Nt, Dd, Nt, Nt, Dd, Eo,
             (long long)Nt * Nt, (long long)Nt * Dd, Hh,
             (long long)Nt * Eo, (long long)Dd, 0, Bsz * Hh);

        // proj + residual -> hY (f32)
        gemm(attbf, wproj + (size_t)l * Eo * Eo, proj_b + (size_t)l * Eo, hX,
             hY, nullptr, Ttok, Eo, Eo, Eo, Eo, Eo, 0, 0, 1, 0, 0, 0, 1);

        k_layernorm<<<Ttok, 256, 0, stream>>>(hY, ln2_s + (size_t)l * Eo, ln2_b + (size_t)l * Eo, lnbf);

        // fc1 + exact GELU -> bf16
        gemm(lnbf, wfc1 + (size_t)l * Eo * Ff, fc1_b + (size_t)l * Ff, nullptr,
             nullptr, mlpbf, Ttok, Ff, Eo, Eo, Ff, Ff, 0, 0, 1, 0, 0, 1, 1);

        // fc2 + residual -> next h (last layer writes d_out)
        float* outp = (l == Ll - 1) ? (float*)d_out : hX;
        gemm(mlpbf, wfc2 + (size_t)l * Ff * Eo, fc2_b + (size_t)l * Eo, hY,
             outp, nullptr, Ttok, Eo, Ff, Ff, Eo, Eo, 0, 0, 1, 0, 0, 0, 1);
    }
}
